// SelfAttention_62354335203902
// MI455X (gfx1250) — matrix operand order repaired and checked by reading, not compiled, not run on hardware
//
#include <hip/hip_runtime.h>
#include <hip/hip_bf16.h>

// ---------------------------------------------------------------------------
// Self-attention block (GroupNorm -> QKV 1x1 -> MHA softmax -> out 1x1 + res)
// B=2, C=256, H=W=64 (N=4096 tokens), 4 heads, head_dim=64, 32 groups.
// Strategy: f32 GN stats, f16 activations/weights, all matmuls through
// v_wmma_f32_16x16x32_f16 (f32 accumulate), flash-attention streaming so the
// 4096x4096 score matrix never touches HBM.  Flash blocks cover 128 q-rows
// (8 waves) so each unique K/V tile is amortized over 2x more WMMA work,
// halving L2-side traffic on the dominant kernel.
// ---------------------------------------------------------------------------

typedef __attribute__((ext_vector_type(16))) _Float16 v16h;
typedef __attribute__((ext_vector_type(8)))  _Float16 v8h;
typedef __attribute__((ext_vector_type(8)))  float    v8f;

#define DI __device__ __forceinline__

static constexpr int Bsz   = 2;
static constexpr int Cch   = 256;
static constexpr int Ntok  = 4096;
static constexpr int NH    = 4;
static constexpr int HD    = 64;
static constexpr int GRP   = 32;
static constexpr int CPG   = Cch / GRP;   // 8

DI v8f zero8() { v8f z = {0.f,0.f,0.f,0.f,0.f,0.f,0.f,0.f}; return z; }

// A-matrix fragment (16x32 f16): lane l holds row m=l&15; hi=l>>4 selects
// 16 contiguous K values {16*hi..16*hi+15}.  rowp points at K=0 of the
// lane's row; k0 is the 32-wide K-step base.
DI v16h load_a_frag(const _Float16* rowp, int k0, int hi) {
  v8h lo = *(const v8h*)(rowp + k0 + 16 * hi);
  v8h hv = *(const v8h*)(rowp + k0 + 16 * hi + 8);
  v16h r;
#pragma unroll
  for (int i = 0; i < 8; ++i) { r[i] = lo[i]; r[i + 8] = hv[i]; }
  return r;
}

// B-matrix fragment (32x16 f16): lane l holds column n=l&15, 16 contiguous
// K values starting at 16*(l>>4).  colp points at K-step base of the column.
DI v16h load_b_frag(const _Float16* colp, int hi) {
  return *(const v16h*)(colp + 16 * hi);
}

DI v8f wmma_f16(v16h a, v16h b, v8f c) {
  return __builtin_amdgcn_wmma_f32_16x16x32_f16(false, a, false, b,
                                                (short)0, c, false, false);
}

DI float red_max16(float v) {
  v = fmaxf(v, __shfl_xor(v, 1, 32));
  v = fmaxf(v, __shfl_xor(v, 2, 32));
  v = fmaxf(v, __shfl_xor(v, 4, 32));
  v = fmaxf(v, __shfl_xor(v, 8, 32));
  return v;
}
DI float red_sum16(float v) {
  v += __shfl_xor(v, 1, 32);
  v += __shfl_xor(v, 2, 32);
  v += __shfl_xor(v, 4, 32);
  v += __shfl_xor(v, 8, 32);
  return v;
}

// ---------------- 1) GroupNorm statistics: one block per (b, group) --------
__global__ void gn_stats(const float* __restrict__ x, float* __restrict__ stats) {
  __shared__ float ssum[256], ssq[256];
  const int bg = blockIdx.x;                         // b*32 + g (contiguous)
  const float* base = x + (size_t)bg * (CPG * Ntok);
  float s = 0.f, q = 0.f;
  for (int i = threadIdx.x; i < CPG * Ntok; i += 256) {
    float v = base[i]; s += v; q += v * v;
  }
  ssum[threadIdx.x] = s; ssq[threadIdx.x] = q;
  __syncthreads();
  for (int st = 128; st > 0; st >>= 1) {
    if (threadIdx.x < st) {
      ssum[threadIdx.x] += ssum[threadIdx.x + st];
      ssq[threadIdx.x]  += ssq[threadIdx.x + st];
    }
    __syncthreads();
  }
  if (threadIdx.x == 0) {
    const float inv = 1.0f / (CPG * Ntok);
    float mean = ssum[0] * inv;
    float var  = ssq[0] * inv - mean * mean;
    stats[bg * 2 + 0] = mean;
    stats[bg * 2 + 1] = rsqrtf(var + 1e-5f);
  }
}

// ---------------- 2) Normalize + affine, write f16 token-major -------------
__global__ void gn_apply(const float* __restrict__ x, const float* __restrict__ stats,
                         const float* __restrict__ sc, const float* __restrict__ bi,
                         _Float16* __restrict__ Xn) {
  size_t idx = (size_t)blockIdx.x * blockDim.x + threadIdx.x;  // b*C*N layout
  int n = (int)(idx & (Ntok - 1));
  int c = (int)((idx >> 12) & (Cch - 1));
  int b = (int)(idx >> 20);
  int bg = b * GRP + (c >> 3);
  float mean = stats[bg * 2 + 0], rstd = stats[bg * 2 + 1];
  float v = (x[idx] - mean) * rstd * sc[c] + bi[c];
  Xn[(((size_t)b << 12) + n) * Cch + c] = (_Float16)v;   // [b][n][c]
}

// ---------------- 3) f32 -> f16 weight conversion --------------------------
__global__ void cvt_w(const float* __restrict__ wq, const float* __restrict__ wo,
                      _Float16* __restrict__ wq16, _Float16* __restrict__ wo16) {
  int i = blockIdx.x * blockDim.x + threadIdx.x;
  if (i < 3 * Cch * Cch) wq16[i] = (_Float16)wq[i];
  if (i < Cch * Cch)     wo16[i] = (_Float16)wo[i];
}

// ---------------- 4) QKV projection GEMM (WMMA), scatter to Q/K/Vt ---------
// D[o, tok] = sum_c W[o,c] * Xn[b][tok][c];  M=768, N=4096, K=256.
__global__ __launch_bounds__(128) void qkv_gemm(
    const _Float16* __restrict__ W, const _Float16* __restrict__ Xn,
    _Float16* __restrict__ Q, _Float16* __restrict__ Kb, _Float16* __restrict__ Vt) {
  const int nb = blockIdx.x, mb = blockIdx.y, b = blockIdx.z;
  const int wave = threadIdx.x >> 5;
  const int lane = threadIdx.x & 31;
  const int li = lane & 15, hi = lane >> 4;
  const int m0 = mb * 64 + wave * 16;
  const int n0 = nb * 64;
  const _Float16* xb = Xn + (size_t)b * Ntok * Cch;
  const _Float16* arow = W + (size_t)(m0 + li) * Cch;

  v8f acc[4];
#pragma unroll
  for (int t = 0; t < 4; ++t) acc[t] = zero8();

  for (int k0 = 0; k0 < Cch; k0 += 32) {
    v16h a = load_a_frag(arow, k0, hi);
#pragma unroll
    for (int t = 0; t < 4; ++t) {
      v16h bf = load_b_frag(xb + (size_t)(n0 + 16 * t + li) * Cch + k0, hi);
      acc[t] = wmma_f16(a, bf, acc[t]);
    }
  }

  const float qs = 0.0625f;  // 1/sqrt(C=256)
#pragma unroll
  for (int t = 0; t < 4; ++t) {
    const int tok = n0 + 16 * t + li;
#pragma unroll
    for (int r = 0; r < 8; ++r) {
      const int o = m0 + r + 8 * hi;
      const int h = o / (3 * HD), rr = o % (3 * HD);
      const size_t bh = (size_t)b * NH + h;
      const float v = acc[t][r];
      if (rr < HD)            Q [(bh * Ntok + tok) * HD + rr]          = (_Float16)(v * qs);
      else if (rr < 2 * HD)   Kb[(bh * Ntok + tok) * HD + (rr - HD)]   = (_Float16)v;
      else                    Vt[(bh * HD + (rr - 2 * HD)) * Ntok + tok] = (_Float16)v;
    }
  }
}

// ---------------- 5) Flash attention: one block per (b, h, 128 q-rows) -----
// 8 waves x 16 q-rows each; K/V tiles are shared across all 8 waves via the
// WGP$ so each unique K/V byte from L2 feeds 2x more WMMA work than with
// 64-row blocks.
__global__ __launch_bounds__(256) void flash_attn(
    const _Float16* __restrict__ Q, const _Float16* __restrict__ Kb,
    const _Float16* __restrict__ Vt, _Float16* __restrict__ Obuf) {
  __shared__ _Float16 Plds[8][16][64];     // per-wave P tile (C-layout -> A-layout)
  const int qt = blockIdx.x, h = blockIdx.y, b = blockIdx.z;
  const int wave = threadIdx.x >> 5;
  const int lane = threadIdx.x & 31;
  const int li = lane & 15, hi = lane >> 4;
  const size_t bh = (size_t)b * NH + h;
  const int q0 = qt * 128 + wave * 16;

  const _Float16* qrow = Q + (bh * Ntok + q0 + li) * HD;
  const v16h aq0 = load_a_frag(qrow, 0, hi);
  const v16h aq1 = load_a_frag(qrow, 32, hi);

  v8f o[4];
#pragma unroll
  for (int t = 0; t < 4; ++t) o[t] = zero8();
  float mrow[8], lrow[8];
#pragma unroll
  for (int r = 0; r < 8; ++r) { mrow[r] = -1e30f; lrow[r] = 0.f; }

  for (int kt = 0; kt < Ntok; kt += 64) {
    // prefetch next k-tile's K/V lines while we compute this one
    if (kt + 64 < Ntok) {
      __builtin_prefetch(Kb + (bh * Ntok + kt + 64 + li) * HD + 16 * hi, 0, 0);
      __builtin_prefetch(Vt + (bh * HD + li) * Ntok + kt + 64 + 16 * hi, 0, 0);
    }
    // ---- S = Q . K^T  (16 q-rows x 64 k-cols) ----
    v8f s[4];
#pragma unroll
    for (int t = 0; t < 4; ++t) {
      const _Float16* kcol = Kb + (bh * Ntok + kt + 16 * t + li) * HD;
      v8f sa = zero8();
      sa = wmma_f16(aq0, load_b_frag(kcol + 0, hi), sa);
      sa = wmma_f16(aq1, load_b_frag(kcol + 32, hi), sa);
      s[t] = sa;
    }
    // ---- online softmax (row stats within each 16-lane half) ----
#pragma unroll
    for (int r = 0; r < 8; ++r) {
      float rm = s[0][r];
#pragma unroll
      for (int t = 1; t < 4; ++t) rm = fmaxf(rm, s[t][r]);
      rm = red_max16(rm);
      const float mn = fmaxf(mrow[r], rm);
      const float scale = __expf(mrow[r] - mn);
      mrow[r] = mn;
      float rs = 0.f;
#pragma unroll
      for (int t = 0; t < 4; ++t) {
        const float p = __expf(s[t][r] - mn);
        rs += p;
        Plds[wave][r + 8 * hi][16 * t + li] = (_Float16)p;
      }
      rs = red_sum16(rs);
      lrow[r] = lrow[r] * scale + rs;
#pragma unroll
      for (int t = 0; t < 4; ++t) o[t][r] *= scale;
    }
    asm volatile("s_wait_dscnt 0x0" ::: "memory");  // wave-private LDS RAW
    const v16h pa0 = load_a_frag(&Plds[wave][li][0], 0, hi);
    const v16h pa1 = load_a_frag(&Plds[wave][li][0], 32, hi);
    // ---- O += P . V  (V stored [b][h][d][tok] -> contiguous k-tokens) ----
#pragma unroll
    for (int t = 0; t < 4; ++t) {
      const _Float16* vrow = Vt + (bh * HD + 16 * t + li) * Ntok + kt;
      o[t] = wmma_f16(pa0, load_b_frag(vrow + 0, hi), o[t]);
      o[t] = wmma_f16(pa1, load_b_frag(vrow + 32, hi), o[t]);
    }
  }

  // ---- normalize by row sums, write token-major f16 [b][tok][c] ----
#pragma unroll
  for (int t = 0; t < 4; ++t) {
    const int c = h * HD + 16 * t + li;
#pragma unroll
    for (int r = 0; r < 8; ++r) {
      const int tok = q0 + r + 8 * hi;
      Obuf[((size_t)b * Ntok + tok) * Cch + c] = (_Float16)(o[t][r] / lrow[r]);
    }
  }
}

// ---------------- 6) Output projection + bias + residual -------------------
__global__ __launch_bounds__(128) void out_gemm(
    const _Float16* __restrict__ W, const _Float16* __restrict__ Ob,
    const float* __restrict__ bias, const float* __restrict__ x,
    float* __restrict__ out) {
  const int nb = blockIdx.x, mb = blockIdx.y, b = blockIdx.z;
  const int wave = threadIdx.x >> 5;
  const int lane = threadIdx.x & 31;
  const int li = lane & 15, hi = lane >> 4;
  const int m0 = mb * 64 + wave * 16;
  const int n0 = nb * 64;
  const _Float16* ob = Ob + (size_t)b * Ntok * Cch;
  const _Float16* arow = W + (size_t)(m0 + li) * Cch;

  v8f acc[4];
#pragma unroll
  for (int t = 0; t < 4; ++t) acc[t] = zero8();

  for (int k0 = 0; k0 < Cch; k0 += 32) {
    v16h a = load_a_frag(arow, k0, hi);
#pragma unroll
    for (int t = 0; t < 4; ++t) {
      v16h bf = load_b_frag(ob + (size_t)(n0 + 16 * t + li) * Cch + k0, hi);
      acc[t] = wmma_f16(a, bf, acc[t]);
    }
  }

#pragma unroll
  for (int t = 0; t < 4; ++t) {
    const int tok = n0 + 16 * t + li;
#pragma unroll
    for (int r = 0; r < 8; ++r) {
      const int oc = m0 + r + 8 * hi;
      const size_t idx = ((size_t)b * Cch + oc) * Ntok + tok;
      out[idx] = acc[t][r] + bias[oc] + x[idx];
    }
  }
}

// ---------------------------------------------------------------------------
extern "C" void kernel_launch(void* const* d_in, const int* in_sizes, int n_in,
                              void* d_out, int out_size, void* d_ws, size_t ws_size,
                              hipStream_t stream) {
  (void)in_sizes; (void)n_in; (void)out_size; (void)ws_size;
  const float* x    = (const float*)d_in[0];
  const float* gsc  = (const float*)d_in[1];
  const float* gbi  = (const float*)d_in[2];
  const float* wqkv = (const float*)d_in[3];
  const float* wout = (const float*)d_in[4];
  const float* bout = (const float*)d_in[5];
  float* out = (float*)d_out;

  char* ws = (char*)d_ws;
  const size_t halfs = 2;
  size_t off = 0;
  float* stats = (float*)(ws + off);               off += 1024;
  _Float16* Wq16 = (_Float16*)(ws + off);          off += (size_t)3 * Cch * Cch * halfs;
  _Float16* Wo16 = (_Float16*)(ws + off);          off += (size_t)Cch * Cch * halfs;
  _Float16* Xn   = (_Float16*)(ws + off);          off += (size_t)Bsz * Ntok * Cch * halfs;
  _Float16* Qb   = (_Float16*)(ws + off);          off += (size_t)Bsz * NH * Ntok * HD * halfs;
  _Float16* Kb   = (_Float16*)(ws + off);          off += (size_t)Bsz * NH * Ntok * HD * halfs;
  _Float16* Vt   = (_Float16*)(ws + off);          off += (size_t)Bsz * NH * HD * Ntok * halfs;
  _Float16* Ob   = (_Float16*)(ws + off);          off += (size_t)Bsz * Ntok * Cch * halfs;

  gn_stats<<<Bsz * GRP, 256, 0, stream>>>(x, stats);
  gn_apply<<<(Bsz * Cch * Ntok) / 256, 256, 0, stream>>>(x, stats, gsc, gbi, Xn);
  cvt_w<<<(3 * Cch * Cch + 255) / 256, 256, 0, stream>>>(wqkv, wout, Wq16, Wo16);
  qkv_gemm<<<dim3(Ntok / 64, (3 * Cch) / 64, Bsz), 128, 0, stream>>>(Wq16, Xn, Qb, Kb, Vt);
  flash_attn<<<dim3(Ntok / 128, NH, Bsz), 256, 0, stream>>>(Qb, Kb, Vt, Ob);
  out_gemm<<<dim3(Ntok / 64, Cch / 64, Bsz), 128, 0, stream>>>(Wo16, Ob, bout, x, out);
}
